// multi_batch_InfoNCE_loss_28767690949114
// MI455X (gfx1250) — matrix operand order repaired and checked by reading, not compiled, not run on hardware
//
#include <hip/hip_runtime.h>
#include <hip/hip_bf16.h>

#define B_   64
#define T_   1024
#define FC_  256
#define FX_  512
#define BT_  (B_ * T_)

typedef __attribute__((ext_vector_type(16))) __bf16       v16bf;
typedef __attribute__((ext_vector_type(8)))  float        v8f;
typedef __attribute__((ext_vector_type(4)))  unsigned int u32x4;

union Frag16 {
    v16bf        v;
    u32x4        q[2];
    unsigned int u[8];
};

static __device__ __forceinline__ unsigned short f32_to_bf16(float f) {
    unsigned int u = __float_as_uint(f);
    u += 0x7FFFu + ((u >> 16) & 1u);          // round-to-nearest-even
    return (unsigned short)(u >> 16);
}

// Packed f32x2 -> bf16x2 (RNE) in a single VALU op: D[15:0]=bf16(lo), D[31:16]=bf16(hi)
static __device__ __forceinline__ unsigned int cvt_pk_bf16(float lo, float hi) {
    unsigned int r;
    asm("v_cvt_pk_bf16_f32 %0, %1, %2" : "=v"(r) : "v"(lo), "v"(hi));
    return r;
}

// ---------------------------------------------------------------------------
// Kernel 0: f32 -> bf16 convert (used for W only; X/C converts are fused)
// ---------------------------------------------------------------------------
__global__ void cvt_bf16_kernel(const float* __restrict__ src,
                                unsigned short* __restrict__ dst, int n) {
    int i = blockIdx.x * blockDim.x + threadIdx.x;
    int stride = gridDim.x * blockDim.x;
    for (; i < n; i += stride) dst[i] = f32_to_bf16(src[i]);
}

// ---------------------------------------------------------------------------
// Kernel 1: Xt[m, c] = sum_k X[m,k] * W[c,k] + b[c]   (m in [0,BT), c in [0,FC))
// Block = 16 rows of X; 4 waves, wave w owns columns [64w, 64w+64).
// A (X rows, f32 -> bf16 fused via v_cvt_pk_bf16_f32), B (W rows, bf16), f32 accum.
// ---------------------------------------------------------------------------
__global__ __launch_bounds__(128)
void gemm_xw_bias_kernel(const float* __restrict__ X,            // [BT, FX] f32
                         const unsigned short* __restrict__ Wb,  // [FC, FX] bf16
                         const float* __restrict__ bias,         // [FC]
                         unsigned short* __restrict__ Xt)        // [BT, FC] bf16
{
    const int lane = threadIdx.x & 31;
    const int wv   = threadIdx.x >> 5;   // 0..3
    const int r16  = lane & 15;
    const int h    = lane >> 4;          // half-wave id
    const int m0   = blockIdx.x << 4;    // 16 rows per block
    const int n0   = wv << 6;            // 64 cols per wave

    v8f acc[4];
    const v8f vzero = {0.f, 0.f, 0.f, 0.f, 0.f, 0.f, 0.f, 0.f};
#pragma unroll
    for (int nt = 0; nt < 4; ++nt) acc[nt] = vzero;

    const float* arow = X + (size_t)(m0 + r16) * FX_;

    for (int kk = 0; kk < FX_; kk += 32) {
        if (kk + 128 < FX_) __builtin_prefetch(arow + kk + 128, 0, 0);

        // A fragment: 16x32 bf16 tile.  Lane (r16,h): k = 8h..8h+7 and 16+8h..16+8h+7
        v8f alo = *(const v8f*)(arow + kk + h * 8);
        v8f ahi = *(const v8f*)(arow + kk + 16 + h * 8);
        Frag16 a;
#pragma unroll
        for (int e = 0; e < 4; ++e) {
            a.u[e]     = cvt_pk_bf16(alo[2 * e], alo[2 * e + 1]);
            a.u[e + 4] = cvt_pk_bf16(ahi[2 * e], ahi[2 * e + 1]);
        }

#pragma unroll
        for (int nt = 0; nt < 4; ++nt) {
            // B fragment: 32x16 bf16, column c = n0+16nt+r16, k = 8h..8h+7 and 16+8h..16+8h+7
            const unsigned short* bcol =
                Wb + (size_t)(n0 + nt * 16 + r16) * FX_ + kk + h * 8;
            Frag16 bfr;
            bfr.q[0] = *(const u32x4*)(bcol);
            bfr.q[1] = *(const u32x4*)(bcol + 16);
            acc[nt] = __builtin_amdgcn_wmma_f32_16x16x32_bf16(
                false, a.v, false, bfr.v, (short)0, acc[nt], false, false);
        }
    }

    // D layout: element (vgpr r, lane): M = r + 8h, N = lane&15
#pragma unroll
    for (int nt = 0; nt < 4; ++nt) {
        const int c = n0 + nt * 16 + r16;
        const float bv = bias[c];
#pragma unroll
        for (int r = 0; r < 8; ++r) {
            const int m = m0 + r + 8 * h;
            Xt[(size_t)m * FC_ + c] = f32_to_bf16(acc[nt][r] + bv);
        }
    }
}

// ---------------------------------------------------------------------------
// Kernel 2: per-timestep Z_t = C_t * Xt_t^T (64x64, K=256), then per-row
// logsumexp over j and diag pick, partial[t] = sum_i (Z[i,i,t] - lse[i,t]).
// Block = one t; wave w owns rows i in [16w, 16w+16), all 64 cols.
// ---------------------------------------------------------------------------
__global__ __launch_bounds__(128)
void z_lse_kernel(const float* __restrict__ C,            // [B, T, FC] f32
                  const unsigned short* __restrict__ Xt,  // [B*T, FC] bf16 (m = j*T + t)
                  float* __restrict__ partial)            // [T]
{
    const int t    = blockIdx.x;
    const int lane = threadIdx.x & 31;
    const int wv   = threadIdx.x >> 5;
    const int r16  = lane & 15;
    const int h    = lane >> 4;

    v8f acc[4];
    const v8f vzero = {0.f, 0.f, 0.f, 0.f, 0.f, 0.f, 0.f, 0.f};
#pragma unroll
    for (int nt = 0; nt < 4; ++nt) acc[nt] = vzero;

    // A row: C[i, t, :], i = 16*wv + r16 (f32, converted on the fly)
    const float* arow = C + ((size_t)(16 * wv + r16) * T_ + t) * FC_;

    for (int kk = 0; kk < FC_; kk += 32) {
        v8f alo = *(const v8f*)(arow + kk + h * 8);
        v8f ahi = *(const v8f*)(arow + kk + 16 + h * 8);
        Frag16 a;
#pragma unroll
        for (int e = 0; e < 4; ++e) {
            a.u[e]     = cvt_pk_bf16(alo[2 * e], alo[2 * e + 1]);
            a.u[e + 4] = cvt_pk_bf16(ahi[2 * e], ahi[2 * e + 1]);
        }

#pragma unroll
        for (int nt = 0; nt < 4; ++nt) {
            // B column j = 16*nt + r16 is row (j*T + t) of Xt, contiguous in c
            const unsigned short* bcol =
                Xt + ((size_t)(nt * 16 + r16) * T_ + t) * FC_ + kk + h * 8;
            Frag16 bfr;
            bfr.q[0] = *(const u32x4*)(bcol);
            bfr.q[1] = *(const u32x4*)(bcol + 16);
            acc[nt] = __builtin_amdgcn_wmma_f32_16x16x32_bf16(
                false, a.v, false, bfr.v, (short)0, acc[nt], false, false);
        }
    }

    // Row i = 16*wv + r + 8h lives on lanes {16h..16h+15}; its 64 values are the
    // 4 n-tile entries at vgpr r.  Reduce max/sumexp across the 16 lanes.
    float rowsum = 0.f;
#pragma unroll
    for (int r = 0; r < 8; ++r) {
        float v0 = acc[0][r], v1 = acc[1][r], v2 = acc[2][r], v3 = acc[3][r];
        float mx = fmaxf(fmaxf(v0, v1), fmaxf(v2, v3));
#pragma unroll
        for (int mask = 1; mask <= 8; mask <<= 1)
            mx = fmaxf(mx, __shfl_xor(mx, mask, 32));
        float s = __expf(v0 - mx) + __expf(v1 - mx) + __expf(v2 - mx) + __expf(v3 - mx);
#pragma unroll
        for (int mask = 1; mask <= 8; mask <<= 1)
            s += __shfl_xor(s, mask, 32);
        const float lse = mx + __logf(s);

        // diag of row i: j = i -> n-tile wv, N = i&15 = r + 8h, held by lane 16h + r + 8h
        if (r16 == r + 8 * h) rowsum += acc[wv][r] - lse;
    }
    // sum the 16 diag-holder lanes of this wave (16 rows)
#pragma unroll
    for (int mask = 1; mask <= 16; mask <<= 1)
        rowsum += __shfl_xor(rowsum, mask, 32);

    __shared__ float wsum[4];
    if (lane == 0) wsum[wv] = rowsum;
    __syncthreads();
    if (threadIdx.x == 0)
        partial[t] = wsum[0] + wsum[1] + wsum[2] + wsum[3];
}

// ---------------------------------------------------------------------------
// Kernel 3: mean over the T_ partials (each partial already sums 64 rows)
// ---------------------------------------------------------------------------
__global__ void reduce_mean_kernel(const float* __restrict__ partial,
                                   float* __restrict__ out) {
    __shared__ float sm[256];
    float s = 0.f;
    for (int i = threadIdx.x; i < T_; i += 256) s += partial[i];
    sm[threadIdx.x] = s;
    __syncthreads();
    for (int off = 128; off > 0; off >>= 1) {
        if ((int)threadIdx.x < off) sm[threadIdx.x] += sm[threadIdx.x + off];
        __syncthreads();
    }
    if (threadIdx.x == 0) out[0] = sm[0] * (1.0f / (float)(B_ * T_));
}

// ---------------------------------------------------------------------------
extern "C" void kernel_launch(void* const* d_in, const int* in_sizes, int n_in,
                              void* d_out, int out_size, void* d_ws, size_t ws_size,
                              hipStream_t stream) {
    (void)in_sizes; (void)n_in; (void)out_size; (void)ws_size;

    const float* C    = (const float*)d_in[0];   // [B, T, FC]
    const float* X    = (const float*)d_in[1];   // [B, T, FX]
    const float* W    = (const float*)d_in[2];   // [FC, FX]
    const float* bias = (const float*)d_in[3];   // [FC]
    float*       out  = (float*)d_out;

    char* ws = (char*)d_ws;
    unsigned short* Wb      = (unsigned short*)(ws);                       // 256 KB
    unsigned short* Xt      = (unsigned short*)(ws + 262144);              // 33.5 MB
    float*          partial = (float*)(ws + 262144 + (size_t)BT_ * FC_ * 2);

    cvt_bf16_kernel<<<256, 256, 0, stream>>>(W, Wb, FC_ * FX_);
    gemm_xw_bias_kernel<<<BT_ / 16, 128, 0, stream>>>(X, Wb, bias, Xt);
    z_lse_kernel<<<T_, 128, 0, stream>>>(C, Xt, partial);
    reduce_mean_kernel<<<1, 256, 0, stream>>>(partial, out);
}
